// W4A8LlamaAttention_81913616270138
// MI455X (gfx1250) — 16-bit and 8-bit matrix operand order repaired and checked by reading, compiled, not run on hardware
//
#include <hip/hip_runtime.h>
#include <cstdint>
#include <cstddef>


typedef __attribute__((ext_vector_type(8)))  int      v8i;
typedef __attribute__((ext_vector_type(8)))  float    v8f;
typedef __attribute__((ext_vector_type(16))) _Float16 v16h;

constexpr int S  = 2048;
constexpr int D  = 4096;
constexpr int H  = 32;
constexpr int HD = 128;


__device__ __forceinline__ v8i load_a_iu8(const int8_t* __restrict__ row,
                                          int k0, int lane) {
  const int kb = (lane >> 4) * 8;
  v8i a;
#pragma unroll
  for (int j = 0; j < 4; ++j) {
    const int* p = (const int*)(row + k0 + kb + 16 * j);
    a[2 * j]     = p[0];
    a[2 * j + 1] = p[1];
  }
  return a;
}

__device__ __forceinline__ v8i load_b_iu8(const int8_t* __restrict__ row,
                                          int k0, int lane) {
  const int kb = (lane >> 4) * 8;
  v8i b;
#pragma unroll
  for (int j = 0; j < 4; ++j) {
    b[2 * j]     = ((const int*)(row + k0 + kb + 16 * j))[0];
    b[2 * j + 1] = ((const int*)(row + k0 + kb + 16 * j))[1];
  }
  return b;
}

__device__ __forceinline__ v8i wmma_iu8(v8i a, v8i b, v8i c) {
  return __builtin_amdgcn_wmma_i32_16x16x64_iu8(true, a, true, b, c, false, false);
}

__device__ __forceinline__ int8_t fquant8(float x, float inv_scale,
                                          float lo, float hi) {
  float r = rintf(x * inv_scale);
  r = fminf(fmaxf(r, lo), hi);
  return (int8_t)(int)r;
}

__global__ void quantize_kernel(const float* __restrict__ x,
                                int8_t* __restrict__ y,
                                float inv_scale, float lo, float hi,
                                size_t n) {
  size_t i = (size_t)blockIdx.x * blockDim.x + threadIdx.x;
  if (i < n) y[i] = fquant8(x[i], inv_scale, lo, hi);
}

__global__ void __launch_bounds__(256)
gemm_iu8_kernel(const int8_t* __restrict__ A, const int8_t* __restrict__ W,
                const float* __restrict__ wscale, float s_in,
                float* __restrict__ out, int M, int N, int K) {
  const int lane = threadIdx.x & 31;
  const int wave = threadIdx.x >> 5;
  const int n0 = blockIdx.x * 128 + wave * 16;
  const int m0 = blockIdx.y * 16;
  const int ml = lane & 15;

  const int8_t* arow = A + (size_t)(m0 + ml) * K;
  const int8_t* brow = W + (size_t)(n0 + ml) * K;

  v8i c = {0, 0, 0, 0, 0, 0, 0, 0};
  for (int k0 = 0; k0 < K; k0 += 64) {
    __builtin_prefetch(brow + k0 + 512, 0, 1);
    v8i a = load_a_iu8(arow, k0, lane);
    v8i b = load_b_iu8(brow, k0, lane);
    c = wmma_iu8(a, b, c);
  }

  const float scl = s_in * wscale[n0 + ml];
  const int mb = (lane >> 4) * 8;
#pragma unroll
  for (int r = 0; r < 8; ++r) {
    out[(size_t)(m0 + mb + r) * N + n0 + ml] = (float)c[r] * scl;
  }
}

__global__ void rope_quant_kernel(const float* __restrict__ qf,
                                  const float* __restrict__ kf,
                                  const float* __restrict__ vf,
                                  const int* __restrict__ pos_ids,
                                  int8_t* __restrict__ q8,
                                  int8_t* __restrict__ k8,
                                  _Float16* __restrict__ vT) {
  const int idx = blockIdx.x * blockDim.x + threadIdx.x;
  const int i = idx & 63;
  const int s = (idx >> 6) & (S - 1);
  const int h = idx >> 17;
  if (h >= H) return;

  const float pos = (float)pos_ids[s];
  const float invf = powf(10000.0f, -((float)i) / 64.0f);
  const float ang = pos * invf;
  const float cs = cosf(ang);
  const float sn = sinf(ang);

  const size_t base = (size_t)s * D + (size_t)h * HD;
  const int j1 = i, j2 = i + 64;
  const float q1 = qf[base + j1], q2 = qf[base + j2];
  const float k1 = kf[base + j1], k2 = kf[base + j2];
  const float v1 = vf[base + j1], v2 = vf[base + j2];

  const float qo1 = q1 * cs - q2 * sn;
  const float qo2 = q2 * cs + q1 * sn;
  const float ko1 = k1 * cs - k2 * sn;
  const float ko2 = k2 * cs + k1 * sn;

  const size_t ob = ((size_t)h * S + s) * HD;
  q8[ob + j1] = fquant8(qo1, 10.0f, -128.0f, 127.0f);
  q8[ob + j2] = fquant8(qo2, 10.0f, -128.0f, 127.0f);
  k8[ob + j1] = fquant8(ko1, 10.0f, -128.0f, 127.0f);
  k8[ob + j2] = fquant8(ko2, 10.0f, -128.0f, 127.0f);

  const int8_t vq1 = fquant8(v1, 10.0f, -128.0f, 127.0f);
  const int8_t vq2 = fquant8(v2, 10.0f, -128.0f, 127.0f);
  vT[((size_t)h * HD + j1) * S + s] = (_Float16)(float)vq1;
  vT[((size_t)h * HD + j2) * S + s] = (_Float16)(float)vq2;
}

__global__ void __launch_bounds__(256)
attn_kernel(const int8_t* __restrict__ q8, const int8_t* __restrict__ k8,
            const _Float16* __restrict__ vT, float* __restrict__ outf) {
  extern __shared__ float smem[];
  float* sc      = smem;
  float* red     = smem + 16 * S;
  float* rowsum  = red + 256;

  const int lane = threadIdx.x & 31;
  const int wave = threadIdx.x >> 5;
  const int q0 = blockIdx.x * 16;
  const int h  = blockIdx.y;
  const int ml = lane & 15;
  const int hi = lane >> 4;

  const int8_t* qrow = q8 + ((size_t)h * S + q0 + ml) * HD;
  const v8i a0 = load_a_iu8(qrow, 0, lane);
  const v8i a1 = load_a_iu8(qrow, 64, lane);
  const float sscale = 0.1f * 0.1f / sqrtf(128.0f);

  for (int kt = wave; kt < S / 16; kt += 8) {
    const int8_t* krow = k8 + ((size_t)h * S + kt * 16 + ml) * HD;
    v8i b0 = load_b_iu8(krow, 0, lane);
    v8i b1 = load_b_iu8(krow, 64, lane);
    v8i cc = {0, 0, 0, 0, 0, 0, 0, 0};
    cc = wmma_iu8(a0, b0, cc);
    cc = wmma_iu8(a1, b1, cc);
    const int kpos = kt * 16 + ml;
#pragma unroll
    for (int r = 0; r < 8; ++r) {
      const int m = r + hi * 8;
      const float mask = (kpos <= q0 + m) ? 0.0f : -1e9f;
      sc[m * S + kpos] = (float)cc[r] * sscale + mask;
    }
  }
  __syncthreads();

  {
    const int row = threadIdx.x & 15;
    const int c0  = threadIdx.x >> 4;
    float pm = -1e30f;
    for (int j = c0; j < S; j += 16) pm = fmaxf(pm, sc[row * S + j]);
    red[row * 16 + c0] = pm;
    __syncthreads();
    float rm = -1e30f;
#pragma unroll
    for (int t = 0; t < 16; ++t) rm = fmaxf(rm, red[row * 16 + t]);
    __syncthreads();
    float ps = 0.0f;
    for (int j = c0; j < S; j += 16) {
      const float e = __expf(sc[row * S + j] - rm);
      sc[row * S + j] = e;
      ps += e;
    }
    red[row * 16 + c0] = ps;
    __syncthreads();
    if (c0 == 0) {
      float t = 0.0f;
#pragma unroll
      for (int u = 0; u < 16; ++u) t += red[row * 16 + u];
      rowsum[row] = t;
    }
  }
  __syncthreads();

  const int kb  = hi * 8;
  const int kb2 = hi * 16;
  const int hd  = wave * 16 + ml;
  const _Float16* vrow = vT + ((size_t)h * HD + hd) * S;

  v8f acc = {0.f, 0.f, 0.f, 0.f, 0.f, 0.f, 0.f, 0.f};
  for (int kt = 0; kt < S / 32; ++kt) {
    v16h pa;
#pragma unroll
    for (int j = 0; j < 8; ++j) {
      pa[j]     = (_Float16)sc[ml * S + kt * 32 + kb + j];
      pa[8 + j] = (_Float16)sc[ml * S + kt * 32 + kb + 16 + j];
    }
    v16h vb;
#pragma unroll
    for (int j = 0; j < 8; ++j) {
      vb[j]     = vrow[kt * 32 + kb + j];
      vb[8 + j] = vrow[kt * 32 + kb + 16 + j];
    }
    acc = __builtin_amdgcn_wmma_f32_16x16x32_f16(false, pa, false, vb,
                                                 (short)0, acc, false, false);
  }

#pragma unroll
  for (int r = 0; r < 8; ++r) {
    const int m = r + hi * 8;
    const float inv = 0.1f / rowsum[m];
    outf[(size_t)(q0 + m) * D + (size_t)h * HD + hd] = acc[r] * inv;
  }
}

extern "C" void kernel_launch(void* const* d_in, const int* in_sizes, int n_in,
                              void* d_out, int out_size, void* d_ws,
                              size_t ws_size, hipStream_t stream) {
  (void)in_sizes; (void)n_in; (void)out_size; (void)ws_size;

  const float* hs = (const float*)d_in[0];
  const float* wq = (const float*)d_in[1];
  const float* wk = (const float*)d_in[2];
  const float* wv = (const float*)d_in[3];
  const float* wo = (const float*)d_in[4];
  const float* sq = (const float*)d_in[5];
  const float* sk = (const float*)d_in[6];
  const float* sv = (const float*)d_in[7];
  const float* so = (const float*)d_in[8];
  const int* pos = (const int*)d_in[10];
  float* out = (float*)d_out;

  char* ws = (char*)d_ws;
  size_t off = 0;
  auto alloc = [&](size_t bytes) -> void* {
    void* p = ws + off;
    off += (bytes + 255) & ~(size_t)255;
    return p;
  };

  const size_t SD = (size_t)S * D;
  const size_t DD = (size_t)D * D;

  int8_t*   act8 = (int8_t*)alloc(SD);
  int8_t*   wq8  = (int8_t*)alloc(DD);
  int8_t*   wk8  = (int8_t*)alloc(DD);
  int8_t*   wv8  = (int8_t*)alloc(DD);
  int8_t*   wo8  = (int8_t*)alloc(DD);
  float*    qf   = (float*)alloc(SD * 4);
  float*    kf   = (float*)alloc(SD * 4);
  float*    vf   = (float*)alloc(SD * 4);
  int8_t*   q8i  = (int8_t*)alloc(SD);
  int8_t*   k8i  = (int8_t*)alloc(SD);
  _Float16* vT   = (_Float16*)alloc(SD * 2);
  float*    attnf = (float*)alloc(SD * 4);
  int8_t*   o8   = (int8_t*)alloc(SD);

  {
    const int blk = 256;
    quantize_kernel<<<(unsigned)((SD + blk - 1) / blk), blk, 0, stream>>>(
        hs, act8, 10.0f, -128.0f, 127.0f, SD);
    const unsigned wgrid = (unsigned)((DD + blk - 1) / blk);
    quantize_kernel<<<wgrid, blk, 0, stream>>>(wq, wq8, 1.0f, -128.0f, 127.0f, DD);
    quantize_kernel<<<wgrid, blk, 0, stream>>>(wk, wk8, 1.0f, -128.0f, 127.0f, DD);
    quantize_kernel<<<wgrid, blk, 0, stream>>>(wv, wv8, 1.0f, -128.0f, 127.0f, DD);
    quantize_kernel<<<wgrid, blk, 0, stream>>>(wo, wo8, 1.0f, -128.0f, 127.0f, DD);
  }

  {
    dim3 grid(D / 128, S / 16);
    gemm_iu8_kernel<<<grid, 256, 0, stream>>>(act8, wq8, sq, 0.1f, qf, S, D, D);
    gemm_iu8_kernel<<<grid, 256, 0, stream>>>(act8, wk8, sk, 0.1f, kf, S, D, D);
    gemm_iu8_kernel<<<grid, 256, 0, stream>>>(act8, wv8, sv, 0.1f, vf, S, D, D);
  }

  {
    const int total = H * S * (HD / 2);
    rope_quant_kernel<<<total / 256, 256, 0, stream>>>(qf, kf, vf, pos,
                                                       q8i, k8i, vT);
  }

  {
    const size_t smem = (size_t)16 * S * 4 + 256 * 4 + 16 * 4;
    dim3 grid(S / 16, H);
    attn_kernel<<<grid, 256, smem, stream>>>(q8i, k8i, vT, attnf);
  }

  {
    const int blk = 256;
    quantize_kernel<<<(unsigned)((SD + blk - 1) / blk), blk, 0, stream>>>(
        attnf, o8, 10.0f, -127.0f, 127.0f, SD);
    dim3 grid(D / 128, S / 16);
    gemm_iu8_kernel<<<grid, 256, 0, stream>>>(o8, wo8, so, 0.1f, out, S, D, D);
  }
}
